// DDSPVocoder_26869315404470
// MI455X (gfx1250) — matrix operand order repaired and checked by reading, not compiled, not run on hardware
//
#include <hip/hip_runtime.h>
#include <math.h>

// ---------------- model constants (match reference) ----------------
#define BATCH   2
#define TFRM    1000
#define HOP     192
#define S_TOT   (TFRM*HOP)        // 192000
#define NHARM   64
#define NNOISE  65
#define IRLEN   128               // 2*(NNOISE-1)
#define CONVL   (HOP+IRLEN)       // 320
#define MLPW    128
#define GH      64
#define FEATW   72                // 64 f0 bins + 8 velocity
#define FEATP   96                // padded K for layer 1 (multiple of 32)
#define HEADN   268               // concat of all head columns
#define HEADP   288               // padded N for head GEMM (multiple of 32)
#define CHUNK   256
#define NCHUNK  (S_TOT/CHUNK)     // 750
#define SPAD    (S_TOT+384)       // overlap-add padded length
#define TWO_PI  6.28318530717958647692f

typedef __attribute__((ext_vector_type(16))) _Float16 v16h;
typedef __attribute__((ext_vector_type(8)))  _Float16 v8h;
typedef __attribute__((ext_vector_type(8)))  float    v8f;

__device__ __forceinline__ float sigmoidf_(float x) { return 1.f / (1.f + __expf(-x)); }
__device__ __forceinline__ float softplusf_(float x) {
    return (x > 20.f) ? x : log1pf(__expf(x));
}

// ---------------- utility kernels ----------------
__global__ void k_zero_f32(float* p, int n) {
    int i = blockIdx.x * blockDim.x + threadIdx.x;
    if (i < n) p[i] = 0.f;
}
// fp32 -> fp16 with row padding K -> Kp (pad region zeroed)
__global__ void k_cvt_pad_f16(const float* s, _Float16* d, int rows, int K, int Kp) {
    int i = blockIdx.x * blockDim.x + threadIdx.x;
    if (i >= rows * Kp) return;
    int r = i / Kp, c = i % Kp;
    d[i] = (c < K) ? (_Float16)s[r * K + c] : (_Float16)0.f;
}
__global__ void k_copy_f32(const float* s, float* d, int n) {
    int i = blockIdx.x * blockDim.x + threadIdx.x;
    if (i < n) d[i] = s[i];
}

// ---------------- feature encoding: f0 soft one-hot + velocity one-hot (padded to 96) --------
__global__ void k_encode_feat(const float* f0, const float* vel, _Float16* feat) {
    int i = blockIdx.x * blockDim.x + threadIdx.x;   // over 2000*96
    if (i >= BATCH * TFRM * FEATP) return;
    int c = i % FEATP, row = i / FEATP;
    int b = row / TFRM, t = row % TFRM;
    float v = 0.f;
    if (c < 64) {
        float f = fmaxf(f0[b * TFRM + t], 20.f);
        float midi = 69.f + 12.f * __log2f(f / 440.f);
        float center = 21.f + (float)c * (87.f / 63.f);
        float sig = 87.f / 64.f;
        float z = (midi - center) / sig;
        v = __expf(-0.5f * z * z);
    } else if (c < FEATW) {
        float vc = fminf(fmaxf(vel[b], 0.f), 7.f);
        int vi = (int)vc;
        v = ((c - 64) == vi) ? 1.f : 0.f;
    }
    feat[i] = (_Float16)v;
}

// ---------------- WMMA GEMM: C[M,N] = act(A[M,K] @ W[N,K]^T + bias) ----------------
// Requirements (guaranteed by caller): M % 16 == 0, N % 32 == 0, K % 32 == 0,
// A/W 16-byte aligned. One wave computes a 16x32 C tile: one A fragment feeds two
// v_wmma_f32_16x16x32_f16 per k-step. All fragment loads are unconditional b128.
// Fragment layouts used here (wave32, same K order in A and B):
//   A: lane%16 = row, lane/16 selects K+16 group; halves = contiguous K run
//   B: lane%16 = col, lane/16 selects K+16 group; halves = contiguous K run
//   C: m = (lane/16)*8 + vgpr, n = lane%16
__global__ void k_gemm_wmma(const _Float16* __restrict__ A, const _Float16* __restrict__ W,
                            const float* __restrict__ bias, float* Cf, _Float16* Ch,
                            int M, int N, int K, int act) {
    int lane = threadIdx.x;                 // 0..31
    int tn = blockIdx.x * 32;
    int tm = blockIdx.y * 16;
    int l15 = lane & 15, lhi = lane >> 4;
    const _Float16* Arow  = A + (size_t)(tm + l15) * K + lhi * 16;
    const _Float16* Wrow0 = W + (size_t)(tn + l15) * K + lhi * 16;
    const _Float16* Wrow1 = Wrow0 + (size_t)16 * K;
    v8f acc0 = {}, acc1 = {};
    for (int k0 = 0; k0 < K; k0 += 32) {
        v8h alo  = *(const v8h*)(Arow + k0);        // K+0..7   (or +16..23 on hi lanes)
        v8h ahi  = *(const v8h*)(Arow + k0 + 8);    // K+8..15  (or +24..31)
        v8h b0lo = *(const v8h*)(Wrow0 + k0);
        v8h b0hi = *(const v8h*)(Wrow0 + k0 + 8);
        v8h b1lo = *(const v8h*)(Wrow1 + k0);
        v8h b1hi = *(const v8h*)(Wrow1 + k0 + 8);
        v16h a  = __builtin_shufflevector(alo,  ahi,  0,1,2,3,4,5,6,7,8,9,10,11,12,13,14,15);
        v16h b0 = __builtin_shufflevector(b0lo, b0hi, 0,1,2,3,4,5,6,7,8,9,10,11,12,13,14,15);
        v16h b1 = __builtin_shufflevector(b1lo, b1hi, 0,1,2,3,4,5,6,7,8,9,10,11,12,13,14,15);
        acc0 = __builtin_amdgcn_wmma_f32_16x16x32_f16(false, a, false, b0, (short)0, acc0, false, false);
        acc1 = __builtin_amdgcn_wmma_f32_16x16x32_f16(false, a, false, b1, (short)0, acc1, false, false);
    }
#pragma unroll
    for (int r = 0; r < 8; ++r) {
        int m  = tm + lhi * 8 + r;
        int n0 = tn + l15;
        int n1 = n0 + 16;
        float v0 = acc0[r] + bias[n0];
        float v1 = acc1[r] + bias[n1];
        if (act == 1) { v0 = fmaxf(v0, 0.f); v1 = fmaxf(v1, 0.f); }
        if (Cf) { Cf[(size_t)m * N + n0] = v0; Cf[(size_t)m * N + n1] = v1; }
        if (Ch) { Ch[(size_t)m * N + n0] = (_Float16)v0; Ch[(size_t)m * N + n1] = (_Float16)v1; }
    }
}

// ---------------- GRU recurrence (sequential), whh resident in LDS ----------------
__global__ void k_gru(const float* __restrict__ xp,   // (B*T, 192) = x@Wih^T + bih
                      const float* __restrict__ whh,  // (192, 64)
                      const float* __restrict__ bhh,  // (192)
                      _Float16* __restrict__ hout) {  // (B*T, 64) fp16
    __shared__ float sW[192 * 64];      // 48 KB of the 320 KB WGP LDS
    __shared__ float sh[BATCH][64];
    __shared__ float shp[BATCH][192];
    int tid = threadIdx.x;              // 384 threads
    for (int i = tid; i < 192 * 64; i += 384) sW[i] = whh[i];
    if (tid < BATCH * 64) sh[tid >> 6][tid & 63] = 0.f;
    __syncthreads();
    int b = tid / 192, g = tid % 192;
    float bg = bhh[g];
    for (int t = 0; t < TFRM; ++t) {
        float acc = bg;
        const float* hb = sh[b];
        const float* wr = &sW[g * 64];
#pragma unroll 8
        for (int j = 0; j < 64; ++j) acc += hb[j] * wr[j];
        shp[b][g] = acc;
        __syncthreads();
        if (tid < BATCH * 64) {
            int bb = tid >> 6, j = tid & 63;
            const float* x = xp + (bb * TFRM + t) * 192;
            float r = sigmoidf_(x[j] + shp[bb][j]);
            float z = sigmoidf_(x[64 + j] + shp[bb][64 + j]);
            float n = tanhf(x[128 + j] + r * shp[bb][128 + j]);
            float h = (1.f - z) * n + z * sh[bb][j];
            sh[bb][j] = h;
            hout[(bb * TFRM + t) * GH + j] = (_Float16)h;
        }
        __syncthreads();
    }
}

// ---------------- head post-process: softmax*softplus amps, sigmoid*softplus noise ----------------
__global__ void k_head_post(const float* __restrict__ HO, float* harmL, float* harmR,
                            float* nzL, float* nzR) {
    __shared__ float red[128];
    int row = blockIdx.x, tid = threadIdx.x;     // 128 threads per (b,t)
    const float* h = HO + (size_t)row * HEADP;
#pragma unroll
    for (int side = 0; side < 2; ++side) {
        float v = (tid < 64) ? h[side * 64 + tid] : -1e30f;
        red[tid] = v; __syncthreads();
        for (int s = 64; s > 0; s >>= 1) { if (tid < s) red[tid] = fmaxf(red[tid], red[tid + s]); __syncthreads(); }
        float mx = red[0]; __syncthreads();
        float e = (tid < 64) ? __expf(v - mx) : 0.f;
        red[tid] = e; __syncthreads();
        for (int s = 64; s > 0; s >>= 1) { if (tid < s) red[tid] += red[tid + s]; __syncthreads(); }
        float sum = red[0]; __syncthreads();
        if (tid < 64) {
            float amp = softplusf_(h[128 + side]);
            float* dst = side ? harmR : harmL;
            dst[(size_t)row * 64 + tid] = e / sum * amp;
        }
        __syncthreads();
    }
    if (tid < NNOISE) {
        float aL = softplusf_(h[260]);
        float aR = softplusf_(h[261]);
        nzL[(size_t)row * NNOISE + tid] = sigmoidf_(h[130 + tid]) * aL;
        nzR[(size_t)row * NNOISE + tid] = sigmoidf_(h[195 + tid]) * aR;
    }
}

// ---------------- per-batch scalar params (means over T) ----------------
__global__ void k_scalars(const float* __restrict__ f0, const float* __restrict__ HO,
                          float* scal) {
    __shared__ float red[256];
    int b = blockIdx.x, tid = threadIdx.x;
    float acc[7] = {0.f, 0.f, 0.f, 0.f, 0.f, 0.f, 0.f};
    for (int t = tid; t < TFRM; t += 256) {
        const float* h = HO + (size_t)(b * TFRM + t) * HEADP;
        acc[0] += fmaxf(f0[b * TFRM + t], 20.f);
        acc[1] += sigmoidf_(h[262]);       // B (inharmonicity, pre-scale)
        acc[2] += softplusf_(h[263]);      // b1f
        acc[3] += softplusf_(h[264]);      // b3f
        acc[4] += softplusf_(h[265]);      // b1s
        acc[5] += softplusf_(h[266]);      // b3s
        acc[6] += sigmoidf_(h[267]);       // alpha
    }
    float res[7];
#pragma unroll
    for (int q = 0; q < 7; ++q) {
        red[tid] = acc[q]; __syncthreads();
        for (int s = 128; s > 0; s >>= 1) { if (tid < s) red[tid] += red[tid + s]; __syncthreads(); }
        res[q] = red[0]; __syncthreads();
    }
    if (tid == 0) {
        float f0m  = res[0] / (float)TFRM;
        float midi = 69.f + 12.f * __log2f(f0m / 440.f);
        float bmax = 0.0008f * __expf(-(midi - 21.f) / 88.f * 2.302585093f);
        scal[b * 8 + 0] = res[1] / (float)TFRM * bmax;   // inh
        scal[b * 8 + 1] = res[2] / (float)TFRM;          // b1f
        scal[b * 8 + 2] = res[3] / (float)TFRM;          // b3f
        scal[b * 8 + 3] = res[4] / (float)TFRM;          // b1s
        scal[b * 8 + 4] = res[5] / (float)TFRM;          // b3s
        scal[b * 8 + 5] = res[6] / (float)TFRM;          // alpha
    }
}

// ---------------- hierarchical cumsum of upsampled f0 ----------------
__device__ __forceinline__ float f0_upsample(const float* f0, int b, int s, int* pi0, int* pi1, float* pw) {
    float pos = fminf(fmaxf((s + 0.5f) * (1.f / HOP) - 0.5f, 0.f), (float)(TFRM - 1));
    int i0 = (int)floorf(pos);
    int i1 = min(i0 + 1, TFRM - 1);
    float w = pos - (float)i0;
    *pi0 = i0; *pi1 = i1; *pw = w;
    return f0[b * TFRM + i0] * (1.f - w) + f0[b * TFRM + i1] * w;
}

__global__ void k_chunk_sums(const float* __restrict__ f0, float* csum) {
    __shared__ float red[CHUNK];
    int c = blockIdx.x, b = blockIdx.y, tid = threadIdx.x;
    int i0, i1; float w;
    float v = f0_upsample(f0, b, c * CHUNK + tid, &i0, &i1, &w);
    red[tid] = v; __syncthreads();
    for (int s = CHUNK / 2; s > 0; s >>= 1) { if (tid < s) red[tid] += red[tid + s]; __syncthreads(); }
    if (tid == 0) csum[b * NCHUNK + c] = red[0];
}

__global__ void k_chunk_prefix(const float* __restrict__ csum, double* cpre) {
    int b = threadIdx.x;
    if (b < BATCH) {
        double a = 0.0;
        for (int c = 0; c < NCHUNK; ++c) { cpre[b * NCHUNK + c] = a; a += (double)csum[b * NCHUNK + c]; }
    }
}

// ---------------- harmonic oscillator bank with two-stage decay ----------------
__global__ void k_harm_synth(const float* __restrict__ f0,
                             const float* __restrict__ harmL, const float* __restrict__ harmR,
                             const float* __restrict__ scal, const double* __restrict__ cpre,
                             float* sigL, float* sigR) {
    __shared__ float ratio[NHARM], sfv[NHARM], ssv[NHARM];
    __shared__ float scan[CHUNK];
    int c = blockIdx.x, b = blockIdx.y, tid = threadIdx.x;   // 256 threads
    float inh = scal[b * 8 + 0], b1f = scal[b * 8 + 1], b3f = scal[b * 8 + 2];
    float b1s = scal[b * 8 + 3], b3s = scal[b * 8 + 4], alpha = scal[b * 8 + 5];
    if (tid < NHARM) {
        float n = (float)tid + 1.f;
        ratio[tid] = n * sqrtf(1.f + inh * n * n);
        sfv[tid] = b1f + b3f * n * n;
        ssv[tid] = b1s + b3s * n * n;
    }
    int s = c * CHUNK + tid;
    int i0, i1; float w;
    float f0u = f0_upsample(f0, b, s, &i0, &i1, &w);
    scan[tid] = f0u; __syncthreads();
    for (int off = 1; off < CHUNK; off <<= 1) {            // inclusive Hillis-Steele scan
        float t2 = (tid >= off) ? scan[tid - off] : 0.f;
        __syncthreads();
        scan[tid] += t2;
        __syncthreads();
    }
    double revs = (cpre[b * NCHUNK + c] + (double)scan[tid]) * (1.0 / 48000.0); // phase in revolutions
    float tsec = (float)s * (1.f / 48000.f);
    const float* hl0 = harmL + (size_t)(b * TFRM + i0) * NHARM;
    const float* hl1 = harmL + (size_t)(b * TFRM + i1) * NHARM;
    const float* hr0 = harmR + (size_t)(b * TFRM + i0) * NHARM;
    const float* hr1 = harmR + (size_t)(b * TFRM + i1) * NHARM;
    float accL = 0.f, accR = 0.f;
    for (int k = 0; k < NHARM; ++k) {
        float rt = ratio[k];
        if (f0u * rt < 24000.f) {                 // anti-alias mask
            double r = revs * (double)rt;
            float ph = (float)(r - floor(r)) * TWO_PI;
            float sv = __sinf(ph);
            float env = alpha * __expf(-sfv[k] * tsec) + (1.f - alpha) * __expf(-ssv[k] * tsec);
            float aL = hl0[k] * (1.f - w) + hl1[k] * w;
            float aR = hr0[k] * (1.f - w) + hr1[k] * w;
            accL += aL * env * sv;
            accR += aR * env * sv;
        }
    }
    sigL[(size_t)b * S_TOT + s] = accL;
    sigR[(size_t)b * S_TOT + s] = accR;
}

// ---------------- noise: analytic irfft(cos basis) + hann, direct 128-tap conv, OLA ----------------
__global__ void k_ir_conv(const float* __restrict__ nzL, const float* __restrict__ nzR,
                          const float* __restrict__ noiseL, const float* __restrict__ noiseR,
                          float* acc) {
    __shared__ float sm[NNOISE];
    __shared__ float sir[IRLEN];
    __shared__ float snz[HOP];
    int t = blockIdx.x, b = blockIdx.y, ch = blockIdx.z, tid = threadIdx.x;  // 128 threads
    const float* nz = (ch ? nzR : nzL) + (size_t)(b * TFRM + t) * NNOISE;
    const float* ns = (ch ? noiseR : noiseL) + (size_t)(b * TFRM + t) * HOP;
    if (tid < NNOISE) sm[tid] = nz[tid];
    snz[tid] = ns[tid];
    if (tid < HOP - IRLEN) snz[IRLEN + tid] = ns[IRLEN + tid];
    __syncthreads();
    {   // ir_rolled[n] * hann[n];  roll folds into (-1)^k on the cosine terms
        int n = tid;
        float a = sm[0] + sm[64] * ((n & 1) ? -1.f : 1.f);
        for (int k = 1; k < 64; ++k) {
            float cs = __cosf(TWO_PI * (float)(k * n) * (1.f / 128.f));
            a += 2.f * sm[k] * ((k & 1) ? -cs : cs);
        }
        a *= (1.f / 128.f);
        float hann = 0.5f - 0.5f * __cosf(TWO_PI * (float)n * (1.f / 128.f));
        sir[n] = a * hann;
    }
    __syncthreads();
    float* dst = acc + (size_t)(b * 2 + ch) * SPAD + (size_t)t * HOP;
    for (int l = tid; l < CONVL; l += 128) {
        int jmin = max(0, l - (HOP - 1));
        int jmax = min(IRLEN - 1, l);
        float a = 0.f;
        for (int j = jmin; j <= jmax; ++j) a += sir[j] * snz[l - j];
        atomicAdd(&dst[l], a);      // adjacent frames overlap by 128 samples
    }
}

// ---------------- final mix with upsampled linear loudness ----------------
__global__ void k_mix(const float* __restrict__ sigL, const float* __restrict__ sigR,
                      const float* __restrict__ acc, const float* __restrict__ ldb,
                      float* out) {
    int idx = blockIdx.x * blockDim.x + threadIdx.x;    // over B*S
    if (idx >= BATCH * S_TOT) return;
    int b = idx / S_TOT, s = idx % S_TOT;
    float pos = fminf(fmaxf((s + 0.5f) * (1.f / HOP) - 0.5f, 0.f), (float)(TFRM - 1));
    int i0 = (int)floorf(pos);
    int i1 = min(i0 + 1, TFRM - 1);
    float w = pos - (float)i0;
    float a0 = __powf(10.f, ldb[b * TFRM + i0] * 0.05f);
    float a1 = __powf(10.f, ldb[b * TFRM + i1] * 0.05f);
    float lo = a0 * (1.f - w) + a1 * w;
    out[(size_t)(b * 2 + 0) * S_TOT + s] = (sigL[(size_t)b * S_TOT + s] + acc[(size_t)(b * 2 + 0) * SPAD + s]) * lo;
    out[(size_t)(b * 2 + 1) * S_TOT + s] = (sigR[(size_t)b * S_TOT + s] + acc[(size_t)(b * 2 + 1) * SPAD + s]) * lo;
}

// ==================================================================
extern "C" void kernel_launch(void* const* d_in, const int* in_sizes, int n_in,
                              void* d_out, int out_size, void* d_ws, size_t ws_size,
                              hipStream_t stream) {
    (void)in_sizes; (void)n_in; (void)out_size; (void)ws_size;
    // setup_inputs() dict order, params flattened in make_params insertion order:
    const float* f0      = (const float*)d_in[0];
    const float* ldb     = (const float*)d_in[1];
    const float* vel     = (const float*)d_in[2];
    const float* pre_w1  = (const float*)d_in[3];
    const float* pre_b1  = (const float*)d_in[4];
    const float* pre_w2  = (const float*)d_in[5];
    const float* pre_b2  = (const float*)d_in[6];
    const float* gru_wih = (const float*)d_in[7];
    const float* gru_whh = (const float*)d_in[8];
    const float* gru_bih = (const float*)d_in[9];
    const float* gru_bhh = (const float*)d_in[10];
    const float* post_w  = (const float*)d_in[11];
    const float* post_b  = (const float*)d_in[12];
    const float* hw[14]  = { (const float*)d_in[13], (const float*)d_in[15],   // harmL_w, harmR_w
                             (const float*)d_in[17], (const float*)d_in[19],   // ampL_w, ampR_w
                             (const float*)d_in[21], (const float*)d_in[23],   // noiseL_w, noiseR_w
                             (const float*)d_in[25], (const float*)d_in[27],   // namL_w, namR_w
                             (const float*)d_in[29], (const float*)d_in[31],   // B_w, b1f_w
                             (const float*)d_in[33], (const float*)d_in[35],   // b3f_w, b1s_w
                             (const float*)d_in[37], (const float*)d_in[39] }; // b3s_w, alpha_w
    const float* hb[14]  = { (const float*)d_in[14], (const float*)d_in[16],
                             (const float*)d_in[18], (const float*)d_in[20],
                             (const float*)d_in[22], (const float*)d_in[24],
                             (const float*)d_in[26], (const float*)d_in[28],
                             (const float*)d_in[30], (const float*)d_in[32],
                             (const float*)d_in[34], (const float*)d_in[36],
                             (const float*)d_in[38], (const float*)d_in[40] };
    const int hrows[14]  = { 64, 64, 1, 1, 65, 65, 1, 1, 1, 1, 1, 1, 1, 1 };   // sum = 268
    const float* noiseL  = (const float*)d_in[41];
    const float* noiseR  = (const float*)d_in[42];

    const int M = BATCH * TFRM;   // 2000 (multiple of 16)

    // ---------------- carve workspace (all blocks 256B aligned) ----------------
    char* base = (char*)d_ws; size_t off = 0;
    auto carve = [&](size_t bytes) -> void* {
        void* p = base + off; off = (off + bytes + 255) & ~(size_t)255; return p;
    };
    _Float16* featH  = (_Float16*)carve((size_t)M * FEATP * 2);        // padded K=96
    _Float16* w1h    = (_Float16*)carve((size_t)MLPW * FEATP * 2);     // padded K=96
    _Float16* w2h    = (_Float16*)carve((size_t)MLPW * MLPW * 2);
    _Float16* wihh   = (_Float16*)carve((size_t)192 * MLPW * 2);
    _Float16* pwh    = (_Float16*)carve((size_t)MLPW * GH * 2);
    _Float16* hWh    = (_Float16*)carve((size_t)HEADP * MLPW * 2);     // rows 268..287 never written
    float*    hB     = (float*)   carve((size_t)HEADP * 4);
    _Float16* h1h    = (_Float16*)carve((size_t)M * MLPW * 2);
    _Float16* h2h    = (_Float16*)carve((size_t)M * MLPW * 2);
    float*    xp     = (float*)   carve((size_t)M * 192 * 4);
    _Float16* hgh    = (_Float16*)carve((size_t)M * GH * 2);
    _Float16* hph    = (_Float16*)carve((size_t)M * MLPW * 2);
    float*    headout= (float*)   carve((size_t)M * HEADP * 4);        // stride 288
    float*    harmLa = (float*)   carve((size_t)M * NHARM * 4);
    float*    harmRa = (float*)   carve((size_t)M * NHARM * 4);
    float*    nzLa   = (float*)   carve((size_t)M * NNOISE * 4);
    float*    nzRa   = (float*)   carve((size_t)M * NNOISE * 4);
    float*    scal   = (float*)   carve((size_t)BATCH * 8 * 4);
    float*    csum   = (float*)   carve((size_t)BATCH * NCHUNK * 4);
    double*   cpre   = (double*)  carve((size_t)BATCH * NCHUNK * 8);
    float*    sigL   = (float*)   carve((size_t)BATCH * S_TOT * 4);
    float*    sigR   = (float*)   carve((size_t)BATCH * S_TOT * 4);
    float*    oacc   = (float*)   carve((size_t)BATCH * 2 * SPAD * 4);

    auto grid1 = [](int n, int bs) { return (n + bs - 1) / bs; };

    // zero overlap-add accumulator (workspace is NOT re-zeroed between replays)
    { int n = BATCH * 2 * SPAD; k_zero_f32<<<grid1(n, 256), 256, 0, stream>>>(oacc, n); }

    // weight conversions fp32 -> fp16 (layer1 K-padded; + concatenated head matrix/bias)
    k_cvt_pad_f16<<<grid1(MLPW * FEATP, 256), 256, 0, stream>>>(pre_w1, w1h, MLPW, FEATW, FEATP);
    k_cvt_pad_f16<<<grid1(MLPW * MLPW, 256), 256, 0, stream>>>(pre_w2, w2h, MLPW, MLPW, MLPW);
    k_cvt_pad_f16<<<grid1(192 * MLPW, 256), 256, 0, stream>>>(gru_wih, wihh, 192, MLPW, MLPW);
    k_cvt_pad_f16<<<grid1(MLPW * GH, 256), 256, 0, stream>>>(post_w, pwh, MLPW, GH, GH);
    {
        int r = 0;
        for (int i = 0; i < 14; ++i) {
            int nW = hrows[i] * MLPW;
            k_cvt_pad_f16<<<grid1(nW, 256), 256, 0, stream>>>(hw[i], hWh + (size_t)r * MLPW,
                                                              hrows[i], MLPW, MLPW);
            k_copy_f32<<<grid1(hrows[i], 64), 64, 0, stream>>>(hb[i], hB + r, hrows[i]);
            r += hrows[i];
        }
        // pad bias rows 268..287 (pad outputs are computed but never read; keep them finite)
        k_zero_f32<<<1, 64, 0, stream>>>(hB + HEADN, HEADP - HEADN);
    }

    // features (K-padded to 96)
    k_encode_feat<<<grid1(M * FEATP, 256), 256, 0, stream>>>(f0, vel, featH);

    // dense stack (WMMA, 16x32 tile per wave): 72(->96)->128 relu, 128->128 relu,
    // 128->192 (GRU input proj), 64->128 relu (post), 128->288 (fused heads, 268 live)
    dim3 wv(32);
    k_gemm_wmma<<<dim3(MLPW / 32, M / 16), wv, 0, stream>>>(featH, w1h, pre_b1, nullptr, h1h, M, MLPW, FEATP, 1);
    k_gemm_wmma<<<dim3(MLPW / 32, M / 16), wv, 0, stream>>>(h1h, w2h, pre_b2, nullptr, h2h, M, MLPW, MLPW, 1);
    k_gemm_wmma<<<dim3(192 / 32, M / 16), wv, 0, stream>>>(h2h, wihh, gru_bih, xp, nullptr, M, 192, MLPW, 0);
    k_gru<<<1, 384, 0, stream>>>(xp, gru_whh, gru_bhh, hgh);
    k_gemm_wmma<<<dim3(MLPW / 32, M / 16), wv, 0, stream>>>(hgh, pwh, post_b, nullptr, hph, M, MLPW, GH, 1);
    k_gemm_wmma<<<dim3(HEADP / 32, M / 16), wv, 0, stream>>>(hph, hWh, hB, headout, nullptr, M, HEADP, MLPW, 0);

    // heads and scalars
    k_head_post<<<M, 128, 0, stream>>>(headout, harmLa, harmRa, nzLa, nzRa);
    k_scalars<<<BATCH, 256, 0, stream>>>(f0, headout, scal);

    // f0 cumsum hierarchy + harmonic synthesis
    k_chunk_sums<<<dim3(NCHUNK, BATCH), CHUNK, 0, stream>>>(f0, csum);
    k_chunk_prefix<<<1, 64, 0, stream>>>(csum, cpre);
    k_harm_synth<<<dim3(NCHUNK, BATCH), CHUNK, 0, stream>>>(f0, harmLa, harmRa, scal, cpre, sigL, sigR);

    // noise: IR synthesis + 128-tap conv + overlap-add
    k_ir_conv<<<dim3(TFRM, BATCH, 2), 128, 0, stream>>>(nzLa, nzRa, noiseL, noiseR, oacc);

    // final mix -> d_out (B,2,S)
    k_mix<<<grid1(BATCH * S_TOT, 256), 256, 0, stream>>>(sigL, sigR, oacc, ldb, (float*)d_out);
}
